// SparseMLP_47278999994761
// MI455X (gfx1250) — compile-verified, matrix operand order checked by reading, not run on hardware
//
#include <hip/hip_runtime.h>
#include <cstdint>
#include <cstddef>

// ---------------------------------------------------------------------------
// MoE MLP for MI455X (gfx1250):  per-expert  out = gelu_tanh(X·W1^T)·W2
//   E=8, T=1024, H=1024, F=4096, fp32 in/out, bf16 WMMA + fp32 accum.
//   Kernel1: g = gelu(X·W1^T)  -> bf16 workspace
//   Kernel2: out = g·W2        (A tile via async global->LDS, ASYNCcnt)
// Block: 256 threads (8 wave32), BM=128 x BN=128, BK=64 (2 WMMA K-steps),
// double-buffered LDS with register-held prefetch of the next K tile.
// ---------------------------------------------------------------------------

typedef __attribute__((ext_vector_type(16))) __bf16 v16bf;
typedef __attribute__((ext_vector_type(2)))  __bf16 v2bf;
typedef __attribute__((ext_vector_type(8)))  float  v8f;
typedef __attribute__((ext_vector_type(4)))  int    v4i;

#define AS1 __attribute__((address_space(1)))
#define AS3 __attribute__((address_space(3)))

#define E_ 8
#define T_ 1024
#define H_ 1024
#define F_ 4096

#define BM 128
#define BN 128
#define BK 64
#define LDSW 72      // padded row stride in halves (144 B, 16-B aligned)

#if __has_builtin(__builtin_amdgcn_global_load_async_to_lds_b128)
#define HAS_ASYNC_LDS 1
#else
#define HAS_ASYNC_LDS 0
#endif

#if HAS_ASYNC_LDS
#if __has_builtin(__builtin_amdgcn_s_wait_asynccnt)
#define WAIT_ASYNC() __builtin_amdgcn_s_wait_asynccnt(0)
#else
#define WAIT_ASYNC() asm volatile("s_wait_asynccnt 0x0" ::: "memory")
#endif
#else
#define WAIT_ASYNC() ((void)0)
#endif

union Frag { v16bf bf; uint4 u[2]; };

// pack two floats -> packed bf16 pair; element writes let clang use the
// native cvt into register halves (no integer shift/or packing)
__device__ __forceinline__ unsigned pk2bf(float a, float b) {
    v2bf v;
    v.x = (__bf16)a;
    v.y = (__bf16)b;
    return __builtin_bit_cast(unsigned, v);
}

__device__ __forceinline__ unsigned short f2bf(float f) {
    return __builtin_bit_cast(unsigned short, (__bf16)f);
}

__device__ __forceinline__ float gelu_tanh(float x) {
    const float c0 = 0.7978845608028654f;   // sqrt(2/pi)
    const float c1 = 0.044715f;
    float u = c0 * (x + c1 * x * x * x);
#if __has_builtin(__builtin_amdgcn_tanhf)
    float t = __builtin_amdgcn_tanhf(u);            // v_tanh_f32 (gfx1250 trans op)
#else
    float e = __expf(2.0f * u);                     // branch-free sigmoid form
    float t = 1.0f - 2.0f * __builtin_amdgcn_rcpf(e + 1.0f);
#endif
    return 0.5f * x * (1.0f + t);
}

// Fragment fetch from LDS (CDNA5 ISA 16-bit A/B 16x32 lane map):
//   lanes 0-15 : row = lane,    K = {kb+0..7, kb+16..23}
//   lanes16-31 : row = lane-16, K = {kb+8..15, kb+24..31}
__device__ __forceinline__ v16bf lds_frag(const unsigned short* base, int row, int kb) {
    Frag f;
    const unsigned short* p = base + row * LDSW + kb;
    f.u[0] = *(const uint4*)(p);
    f.u[1] = *(const uint4*)(p + 16);
    return f.bf;
}

// ---------------------------------------------------------------------------
// Kernel 1: g[e] = gelu(X[e] (T x H)  ·  W1[e]^T (H x F))   -> bf16
// grid = (F/BN, T/BM, E), block = 256
// ---------------------------------------------------------------------------
__global__ __launch_bounds__(256)
void moe_gemm1_gelu(const float* __restrict__ x,
                    const float* __restrict__ w1,
                    unsigned short* __restrict__ g)
{
    __shared__ unsigned short As[2][BM * LDSW];
    __shared__ unsigned short Bs[2][BN * LDSW];

    const int e  = blockIdx.z;
    const int bx = blockIdx.x;          // F tile
    const int by = blockIdx.y;          // T tile
    const int tid  = threadIdx.x;
    const int lane = tid & 31;
    const int wave = tid >> 5;
    const int wm = wave & 1;            // 2 row groups of 64
    const int wn = wave >> 1;           // 4 col groups of 32

    const float* xe  = x  + (size_t)e * T_ * H_ + (size_t)(by * BM) * H_;
    const float* w1e = w1 + (size_t)e * F_ * H_ + (size_t)(bx * BN) * H_;
    unsigned short* ge = g + (size_t)e * T_ * F_;

    v8f acc[4][2];
    #pragma unroll
    for (int mt = 0; mt < 4; ++mt)
        #pragma unroll
        for (int nt = 0; nt < 2; ++nt)
            acc[mt][nt] = (v8f)(0.0f);

    const int nlocal = lane & 15;
    const int kbase  = (lane >> 4) * 8;
    const int NK = H_ / BK;             // 16

    float4 pfA[8], pfB[8];

    // ---- prologue: tile 0 -> buffer 0
    #pragma unroll
    for (int i = 0; i < 8; ++i) {
        int slot = tid + 256 * i, row = slot >> 4, c4 = slot & 15;
        pfA[i] = *(const float4*)(xe  + (size_t)row * H_ + c4 * 4);
        pfB[i] = *(const float4*)(w1e + (size_t)row * H_ + c4 * 4);
    }
    #pragma unroll
    for (int i = 0; i < 8; ++i) {
        int slot = tid + 256 * i, row = slot >> 4, c4 = slot & 15;
        *(uint2*)(&As[0][row * LDSW + c4 * 4]) =
            make_uint2(pk2bf(pfA[i].x, pfA[i].y), pk2bf(pfA[i].z, pfA[i].w));
        *(uint2*)(&Bs[0][row * LDSW + c4 * 4]) =
            make_uint2(pk2bf(pfB[i].x, pfB[i].y), pk2bf(pfB[i].z, pfB[i].w));
    }
    __syncthreads();

    for (int k = 0; k < NK; ++k) {
        const int cur = k & 1;
        const int nxt = cur ^ 1;
        const bool hasNext = (k + 1) < NK;
        const int k0n = (k + 1) * BK;

        // issue next tile's global loads (latency hidden under the WMMAs)
        if (hasNext) {
            #pragma unroll
            for (int i = 0; i < 8; ++i) {
                int slot = tid + 256 * i, row = slot >> 4, c4 = slot & 15;
                pfA[i] = *(const float4*)(xe  + (size_t)row * H_ + k0n + c4 * 4);
                pfB[i] = *(const float4*)(w1e + (size_t)row * H_ + k0n + c4 * 4);
            }
        }

        // compute on current buffer
        #pragma unroll
        for (int ks = 0; ks < BK; ks += 32) {
            int kb = ks + kbase;
            v16bf aF[4], bF[2];
            #pragma unroll
            for (int mt = 0; mt < 4; ++mt)
                aF[mt] = lds_frag(&As[cur][0], wm * 64 + mt * 16 + nlocal, kb);
            #pragma unroll
            for (int nt = 0; nt < 2; ++nt)
                bF[nt] = lds_frag(&Bs[cur][0], wn * 32 + nt * 16 + nlocal, kb);

            #pragma unroll
            for (int mt = 0; mt < 4; ++mt)
                #pragma unroll
                for (int nt = 0; nt < 2; ++nt)
                    acc[mt][nt] = __builtin_amdgcn_wmma_f32_16x16x32_bf16(
                        false, aF[mt], false, bF[nt],
                        (short)0, acc[mt][nt], false, false);
        }

        // convert + store next tile into the other buffer
        if (hasNext) {
            #pragma unroll
            for (int i = 0; i < 8; ++i) {
                int slot = tid + 256 * i, row = slot >> 4, c4 = slot & 15;
                *(uint2*)(&As[nxt][row * LDSW + c4 * 4]) =
                    make_uint2(pk2bf(pfA[i].x, pfA[i].y), pk2bf(pfA[i].z, pfA[i].w));
                *(uint2*)(&Bs[nxt][row * LDSW + c4 * 4]) =
                    make_uint2(pk2bf(pfB[i].x, pfB[i].y), pk2bf(pfB[i].z, pfB[i].w));
            }
        }
        __syncthreads();
    }

    // epilogue: GELU + bf16 store.  C layout: lanes 0-15 -> M=r, lanes16-31 -> M=r+8
    const int rowOff = by * BM + wm * 64;
    const int colOff = bx * BN + wn * 32;
    const int rplus  = (lane >> 4) * 8;
    #pragma unroll
    for (int mt = 0; mt < 4; ++mt)
        #pragma unroll
        for (int nt = 0; nt < 2; ++nt) {
            int col = colOff + nt * 16 + nlocal;
            #pragma unroll
            for (int r = 0; r < 8; ++r) {
                int row = rowOff + mt * 16 + rplus + r;
                ge[(size_t)row * F_ + col] = f2bf(gelu_tanh(acc[mt][nt][r]));
            }
        }
}

// ---------------------------------------------------------------------------
// Kernel 2: out[e] = g[e] (T x F, bf16)  ·  W2[e] (F x H, fp32)
// grid = (H/BN, T/BM, E), block = 256
// ---------------------------------------------------------------------------
__global__ __launch_bounds__(256)
void moe_gemm2(const unsigned short* __restrict__ g,
               const float* __restrict__ w2,
               float* __restrict__ out)
{
    __shared__ unsigned short As[2][BM * LDSW];
    __shared__ unsigned short Bs[2][BN * LDSW];

    const int e  = blockIdx.z;
    const int bx = blockIdx.x;          // H tile
    const int by = blockIdx.y;          // T tile
    const int tid  = threadIdx.x;
    const int lane = tid & 31;
    const int wave = tid >> 5;
    const int wm = wave & 1;
    const int wn = wave >> 1;

    const unsigned short* ge = g + (size_t)e * T_ * F_ + (size_t)(by * BM) * F_;
    const float* w2e = w2 + (size_t)e * F_ * H_;
    float* oute = out + (size_t)e * T_ * H_;

    v8f acc[4][2];
    #pragma unroll
    for (int mt = 0; mt < 4; ++mt)
        #pragma unroll
        for (int nt = 0; nt < 2; ++nt)
            acc[mt][nt] = (v8f)(0.0f);

    const int nlocal = lane & 15;
    const int kbase  = (lane >> 4) * 8;
    const int NK = F_ / BK;             // 64

    float4 pfB0[4], pfB1[4];
#if !HAS_ASYNC_LDS
    uint4 pfAu[4];
#endif

    // A tile staging: async global->LDS 16B copies (8 x 16B per 64-half row)
    auto issueA = [&](int k0, int buf) {
        #pragma unroll
        for (int i = 0; i < 4; ++i) {
            int slot = tid + 256 * i, row = slot >> 3, c4 = slot & 7;
            const unsigned short* pa = ge + (size_t)row * F_ + k0 + c4 * 8;
#if HAS_ASYNC_LDS
            __builtin_amdgcn_global_load_async_to_lds_b128(
                (AS1 v4i*)(uintptr_t)pa,
                (AS3 v4i*)(unsigned)(uintptr_t)(&As[buf][row * LDSW + c4 * 8]),
                0, 0);
#else
            pfAu[i] = *(const uint4*)pa;
#endif
        }
    };
#if !HAS_ASYNC_LDS
    auto storeA = [&](int buf) {
        #pragma unroll
        for (int i = 0; i < 4; ++i) {
            int slot = tid + 256 * i, row = slot >> 3, c4 = slot & 7;
            *(uint4*)(&As[buf][row * LDSW + c4 * 8]) = pfAu[i];
        }
    };
#endif
    // B tile: W2 [K rows, N contiguous]; load two adjacent K rows per slot
    auto loadB = [&](int k0) {
        #pragma unroll
        for (int i = 0; i < 4; ++i) {
            int slot = tid + 256 * i, kp = slot >> 5, c4 = slot & 31;
            const float* pb0 = w2e + (size_t)(k0 + 2 * kp) * H_ + bx * BN + c4 * 4;
            pfB0[i] = *(const float4*)pb0;
            pfB1[i] = *(const float4*)(pb0 + H_);
        }
    };
    // transpose into Bs[n][k]: pack (k, k+1) bf16 pair per 32-bit LDS store
    auto storeB = [&](int buf) {
        #pragma unroll
        for (int i = 0; i < 4; ++i) {
            int slot = tid + 256 * i, kp = slot >> 5, c4 = slot & 31;
            float a0[4] = { pfB0[i].x, pfB0[i].y, pfB0[i].z, pfB0[i].w };
            float a1[4] = { pfB1[i].x, pfB1[i].y, pfB1[i].z, pfB1[i].w };
            #pragma unroll
            for (int j = 0; j < 4; ++j) {
                int n = c4 * 4 + j;
                *(unsigned*)(&Bs[buf][n * LDSW + 2 * kp]) = pk2bf(a0[j], a1[j]);
            }
        }
    };

    // ---- prologue: tile 0 -> buffer 0
    issueA(0, 0);
    loadB(0);
#if !HAS_ASYNC_LDS
    storeA(0);
#endif
    storeB(0);
    WAIT_ASYNC();
    __syncthreads();

    for (int k = 0; k < NK; ++k) {
        const int cur = k & 1;
        const int nxt = cur ^ 1;
        const bool hasNext = (k + 1) < NK;
        const int k0n = (k + 1) * BK;

        if (hasNext) {
            issueA(k0n, nxt);       // async copies straight into LDS, no VGPRs
            loadB(k0n);             // fp32 loads held in registers
        }

        #pragma unroll
        for (int ks = 0; ks < BK; ks += 32) {
            int kb = ks + kbase;
            v16bf aF[4], bF[2];
            #pragma unroll
            for (int mt = 0; mt < 4; ++mt)
                aF[mt] = lds_frag(&As[cur][0], wm * 64 + mt * 16 + nlocal, kb);
            #pragma unroll
            for (int nt = 0; nt < 2; ++nt)
                bF[nt] = lds_frag(&Bs[cur][0], wn * 32 + nt * 16 + nlocal, kb);

            #pragma unroll
            for (int mt = 0; mt < 4; ++mt)
                #pragma unroll
                for (int nt = 0; nt < 2; ++nt)
                    acc[mt][nt] = __builtin_amdgcn_wmma_f32_16x16x32_bf16(
                        false, aF[mt], false, bF[nt],
                        (short)0, acc[mt][nt], false, false);
        }

        if (hasNext) {
#if !HAS_ASYNC_LDS
            storeA(nxt);
#endif
            storeB(nxt);
            WAIT_ASYNC();
        }
        __syncthreads();
    }

    const int rowOff = by * BM + wm * 64;
    const int colOff = bx * BN + wn * 32;
    const int rplus  = (lane >> 4) * 8;
    #pragma unroll
    for (int mt = 0; mt < 4; ++mt)
        #pragma unroll
        for (int nt = 0; nt < 2; ++nt) {
            int col = colOff + nt * 16 + nlocal;
            #pragma unroll
            for (int r = 0; r < 8; ++r) {
                int row = rowOff + mt * 16 + rplus + r;
                oute[(size_t)row * H_ + col] = acc[mt][nt][r];
            }
        }
}

// ---------------------------------------------------------------------------
extern "C" void kernel_launch(void* const* d_in, const int* in_sizes, int n_in,
                              void* d_out, int out_size, void* d_ws, size_t ws_size,
                              hipStream_t stream) {
    const float* x  = (const float*)d_in[0];   // [E,T,H]
    const float* w1 = (const float*)d_in[1];   // [E,F,H]
    const float* w2 = (const float*)d_in[2];   // [E,F,H]
    float* out = (float*)d_out;                // [E,T,H] fp32
    unsigned short* g = (unsigned short*)d_ws; // [E,T,F] bf16 (64 MB)

    dim3 blk(256, 1, 1);
    dim3 g1(F_ / BN, T_ / BM, E_);
    dim3 g2(H_ / BN, T_ / BM, E_);
    hipLaunchKernelGGL(moe_gemm1_gelu, g1, blk, 0, stream, x, w1, g);
    hipLaunchKernelGGL(moe_gemm2,      g2, blk, 0, stream, g, w2, out);
}
